// PLRNN_44762149159376
// MI455X (gfx1250) — matrix operand order repaired and checked by reading, not compiled, not run on hardware
//
#include <hip/hip_runtime.h>

typedef __attribute__((ext_vector_type(16))) __bf16 bf16x16;
typedef __attribute__((ext_vector_type(8)))  float  f32x8;

#define DEV static __device__ __forceinline__

constexpr int B_  = 256;
constexpr int T_  = 1024;
constexpr int DX_ = 64;
constexpr int DZ_ = 128;
constexpr int DB_ = 16;
constexpr int THP_ = DB_ + 1;   // padded theta row stride (bank-conflict free)
constexpr int TAU_ = 25;
constexpr float CLIP_ = 10.0f;

// Hardware tanh on gfx1250 (v_tanh_f32, single TRANS op) with safe fallback.
#if __has_builtin(__builtin_amdgcn_tanhf)
DEV float fast_tanh(float x) { return __builtin_amdgcn_tanhf(x); }
#elif __has_builtin(__builtin_amdgcn_tanh_f32)
DEV float fast_tanh(float x) { return __builtin_amdgcn_tanh_f32(x); }
#else
DEV float fast_tanh(float x) { return tanhf(x); }
#endif

// ---------- WMMA fragment helpers (wave32, 16x16x32 bf16) ----------

// A-matrix 16x32 (MxK) bf16 fragment from LDS row-major [16][ldk].
// lane l<16: M=l, elems 0..15 -> K=k0..k0+15
// lane l>=16: M=l-16, elems 0..15 -> K=k0+16..31
DEV bf16x16 load_a_frag(const __bf16* m, int ldk, int k0, int lane) {
  const int r  = lane & 15;
  const int hf = lane >> 4;
  const __bf16* row = m + r * ldk + k0 + hf * 16;
  bf16x16 a;
#pragma unroll
  for (int i = 0; i < 8; ++i) a[i] = row[i];
#pragma unroll
  for (int i = 0; i < 8; ++i) a[8 + i] = row[8 + i];
  return a;
}

// B-matrix 32x16 (KxN) fragment where B[k][n] = W[n][k], W global row-major
// [nrows][ld] f32.  lanes 0..15: K=k0..k0+15, lanes 16..31: K=k0+16..31, N=lane&15.
// zero_diag: implements W = AW - diag(AW).
DEV bf16x16 load_b_frag_w(const float* W, int ld, int n0, int k0, int lane,
                          bool zero_diag) {
  const int n  = n0 + (lane & 15);
  const int kb = k0 + (lane >> 4) * 16;
  const float* row = W + (size_t)n * ld + kb;
  bf16x16 b;
#pragma unroll
  for (int i = 0; i < 16; ++i) {
    float v = row[i];
    if (zero_diag && (kb + i) == n) v = 0.0f;
    b[i] = (__bf16)v;
  }
  return b;
}

DEV f32x8 wmma_bf16(bf16x16 a, bf16x16 b, f32x8 c) {
  return __builtin_amdgcn_wmma_f32_16x16x32_bf16(false, a, false, b, (short)0,
                                                 c, false, false);
}

// ---------- Encoder: only the 40 teacher-forcing timesteps ----------
// Writes tanh(tanh(x@w1^T+b1)@w2^T+b2) into the z_seq region of d_out at
// (b, t=TAU*(i+1)) slots; the scan kernel consumes then overwrites them.
__global__ __launch_bounds__(256) void enc_kernel(
    const float* __restrict__ x,                     // (B,T,DX)
    const float* __restrict__ w1, const float* __restrict__ b1,  // (DZ,DX),(DZ)
    const float* __restrict__ w2, const float* __restrict__ b2,  // (DZ,DZ),(DZ)
    float* __restrict__ outz)                        // (B,T,DZ)
{
  __shared__ __bf16 xs[16 * DX_];
  __shared__ __bf16 h1[16 * DZ_];

  const int tid  = threadIdx.x;
  const int lane = tid & 31;
  const int wave = tid >> 5;
  const int hf   = lane >> 4;

  const int tfi   = blockIdx.x >> 4;  // 0..39
  const int btile = blockIdx.x & 15;  // 0..15
  const int t  = TAU_ * (tfi + 1);
  const int b0 = btile * 16;

  for (int i = tid; i < 16 * DX_; i += 256) {
    const int m = i >> 6, k = i & 63;
    xs[i] = (__bf16)x[((size_t)(b0 + m) * T_ + t) * DX_ + k];
  }

  bf16x16 w1f[2], w2f[4];
#pragma unroll
  for (int kc = 0; kc < 2; ++kc)
    w1f[kc] = load_b_frag_w(w1, DX_, wave * 16, kc * 32, lane, false);
#pragma unroll
  for (int kc = 0; kc < 4; ++kc)
    w2f[kc] = load_b_frag_w(w2, DZ_, wave * 16, kc * 32, lane, false);

  const int   ncol  = wave * 16 + (lane & 15);
  const float bias1 = b1[ncol];
  const float bias2 = b2[ncol];

  __syncthreads();

  f32x8 acc = {};
#pragma unroll
  for (int kc = 0; kc < 2; ++kc)
    acc = wmma_bf16(load_a_frag(xs, DX_, kc * 32, lane), w1f[kc], acc);

#pragma unroll
  for (int v = 0; v < 8; ++v) {
    const int m = v + 8 * hf;
    h1[m * DZ_ + ncol] = (__bf16)fast_tanh(acc[v] + bias1);
  }
  __syncthreads();

  f32x8 acc2 = {};
#pragma unroll
  for (int kc = 0; kc < 4; ++kc)
    acc2 = wmma_bf16(load_a_frag(h1, DZ_, kc * 32, lane), w2f[kc], acc2);

#pragma unroll
  for (int v = 0; v < 8; ++v) {
    const int m = v + 8 * hf;
    outz[((size_t)(b0 + m) * T_ + t) * DZ_ + ncol] = fast_tanh(acc2[v] + bias2);
  }
}

// ---------- Recurrent scan: 16 persistent blocks, each owns 16 batch rows ----------
__global__ __launch_bounds__(256) void scan_kernel(
    const float* __restrict__ z0,      // (B,DZ)
    const float* __restrict__ AW,      // (DZ,DZ)
    const float* __restrict__ h,       // (DZ)
    const float* __restrict__ alphas,  // (DB)
    const float* __restrict__ thetas,  // (DZ,DB)
    float* __restrict__ outz)          // (B,T,DZ)
{
  __shared__ float  zf[16 * DZ_];     // current z state (f32)
  __shared__ __bf16 basis[16 * DZ_];  // WMMA A operand staging
  __shared__ float  th[DZ_ * THP_];   // padded: bank-conflict-free inner loop
  __shared__ float  al[DB_];

  const int tid  = threadIdx.x;
  const int lane = tid & 31;
  const int wave = tid >> 5;
  const int hf   = lane >> 4;
  const int b0   = blockIdx.x * 16;

  for (int i = tid; i < DZ_ * DB_; i += 256)
    th[(i >> 4) * THP_ + (i & 15)] = thetas[i];
  if (tid < DB_) al[tid] = alphas[tid];
  for (int i = tid; i < 16 * DZ_; i += 256) {
    const int m = i >> 7, n = i & 127;
    zf[i] = z0[(size_t)(b0 + m) * DZ_ + n];
  }

  // W^T fragments (B[k][n] = AW[n][k], diagonal zeroed), resident in VGPRs.
  bf16x16 wf[4];
#pragma unroll
  for (int kc = 0; kc < 4; ++kc)
    wf[kc] = load_b_frag_w(AW, DZ_, wave * 16, kc * 32, lane, true);

  const int   ncol  = wave * 16 + (lane & 15);
  const float adiag = AW[(size_t)ncol * DZ_ + ncol];  // A = diag(AW)
  const float hbias = h[ncol];

  const int row = tid >> 4;  // 0..15 (two rows per wave)
  const int sub = tid & 15;

  __syncthreads();

  for (int t = 0; t < T_; ++t) {
    if (t > 0 && (t % TAU_) == 0) {  // teacher forcing (slots written by enc_kernel)
      for (int i = tid; i < 16 * DZ_; i += 256) {
        const int m = i >> 7, n = i & 127;
        zf[i] = outz[((size_t)(b0 + m) * T_ + t) * DZ_ + n];
      }
      __syncthreads();
    }

    // per-row mean over DZ (16 lanes per row, 8 elems each, shfl-xor reduce)
    float s = 0.f;
#pragma unroll
    for (int j = 0; j < 8; ++j) s += zf[row * DZ_ + sub + 16 * j];
#pragma unroll
    for (int o = 8; o >= 1; o >>= 1) s += __shfl_xor(s, o, 32);
    const float mean = s * (1.0f / DZ_);

    // basis[b][k] = sum_d alphas[d] * relu(zc + thetas[k][d])
#pragma unroll
    for (int j = 0; j < 8; ++j) {
      const int   k  = sub + 16 * j;
      const float zc = zf[row * DZ_ + k] - mean;
      float a = 0.f;
#pragma unroll
      for (int d = 0; d < DB_; ++d)
        a += al[d] * fmaxf(zc + th[k * THP_ + d], 0.f);
      basis[row * DZ_ + k] = (__bf16)a;
    }
    __syncthreads();

    // z_next = A*z + basis @ W^T + h, clipped
    f32x8 acc = {};
#pragma unroll
    for (int kc = 0; kc < 4; ++kc)
      acc = wmma_bf16(load_a_frag(basis, DZ_, kc * 32, lane), wf[kc], acc);

#pragma unroll
    for (int v = 0; v < 8; ++v) {
      const int m = v + 8 * hf;
      float zn = adiag * zf[m * DZ_ + ncol] + acc[v] + hbias;
      zn = fminf(fmaxf(zn, -CLIP_), CLIP_);
      outz[((size_t)(b0 + m) * T_ + t) * DZ_ + ncol] = zn;
      zf[m * DZ_ + ncol] = zn;  // (m,ncol) owned exclusively by this lane
    }
    __syncthreads();
  }
}

// ---------- Decoder: 2-layer tanh MLP over all B*T rows ----------
__global__ __launch_bounds__(256) void dec_kernel(
    const float* __restrict__ zin,                   // (B*T, DZ)
    const float* __restrict__ w1, const float* __restrict__ b1,  // (DZ,DZ),(DZ)
    const float* __restrict__ w2, const float* __restrict__ b2,  // (DX,DZ),(DX)
    float* __restrict__ xout)                        // (B*T, DX)
{
  __shared__ __bf16 zs[16 * DZ_];
  __shared__ __bf16 h1[16 * DZ_];

  const int tid  = threadIdx.x;
  const int lane = tid & 31;
  const int wave = tid >> 5;
  const int hf   = lane >> 4;
  const size_t r0 = (size_t)blockIdx.x * 16;

  for (int i = tid; i < 16 * DZ_; i += 256) {
    const int m = i >> 7, k = i & 127;
    zs[i] = (__bf16)zin[(r0 + m) * DZ_ + k];
  }

  bf16x16 w1f[4];
#pragma unroll
  for (int kc = 0; kc < 4; ++kc)
    w1f[kc] = load_b_frag_w(w1, DZ_, wave * 16, kc * 32, lane, false);

  const int   ncol  = wave * 16 + (lane & 15);
  const float bias1 = b1[ncol];

  __syncthreads();

  f32x8 acc = {};
#pragma unroll
  for (int kc = 0; kc < 4; ++kc)
    acc = wmma_bf16(load_a_frag(zs, DZ_, kc * 32, lane), w1f[kc], acc);

#pragma unroll
  for (int v = 0; v < 8; ++v) {
    const int m = v + 8 * hf;
    h1[m * DZ_ + ncol] = (__bf16)fast_tanh(acc[v] + bias1);
  }
  __syncthreads();

  // layer 2: N = DX = 64 -> 4 N-tiles handled by waves 0..3 (wave-uniform branch)
  if (wave < 4) {
    bf16x16 w2f[4];
#pragma unroll
    for (int kc = 0; kc < 4; ++kc)
      w2f[kc] = load_b_frag_w(w2, DZ_, wave * 16, kc * 32, lane, false);
    const int   nc2   = wave * 16 + (lane & 15);
    const float bias2 = b2[nc2];

    f32x8 acc2 = {};
#pragma unroll
    for (int kc = 0; kc < 4; ++kc)
      acc2 = wmma_bf16(load_a_frag(h1, DZ_, kc * 32, lane), w2f[kc], acc2);

#pragma unroll
    for (int v = 0; v < 8; ++v) {
      const int m = v + 8 * hf;
      xout[(r0 + m) * DX_ + nc2] = fast_tanh(acc2[v] + bias2);
    }
  }
}

extern "C" void kernel_launch(void* const* d_in, const int* in_sizes, int n_in,
                              void* d_out, int out_size, void* d_ws, size_t ws_size,
                              hipStream_t stream) {
  const float* x      = (const float*)d_in[0];
  const float* z0     = (const float*)d_in[1];
  const float* AW     = (const float*)d_in[2];
  const float* h      = (const float*)d_in[3];
  const float* alphas = (const float*)d_in[4];
  const float* thetas = (const float*)d_in[5];
  const float* enc_w1 = (const float*)d_in[6];
  const float* enc_b1 = (const float*)d_in[7];
  const float* enc_w2 = (const float*)d_in[8];
  const float* enc_b2 = (const float*)d_in[9];
  const float* dec_w1 = (const float*)d_in[10];
  const float* dec_b1 = (const float*)d_in[11];
  const float* dec_w2 = (const float*)d_in[12];
  const float* dec_b2 = (const float*)d_in[13];

  float* xpred = (float*)d_out;                       // (B,T,DX)
  float* zseq  = xpred + (size_t)B_ * T_ * DX_;       // (B,T,DZ)

  // 40 teacher-forcing timesteps x 16 batch tiles
  enc_kernel<<<dim3(40 * 16), dim3(256), 0, stream>>>(x, enc_w1, enc_b1,
                                                      enc_w2, enc_b2, zseq);
  // 16 persistent blocks, one per 16-row batch tile, scan all T steps
  scan_kernel<<<dim3(16), dim3(256), 0, stream>>>(z0, AW, h, alphas, thetas,
                                                  zseq);
  // (B*T)/16 row tiles
  dec_kernel<<<dim3((B_ * T_) / 16), dim3(256), 0, stream>>>(
      zseq, dec_w1, dec_b1, dec_w2, dec_b2, xpred);
}
